// TimeLSTM_39530878992939
// MI455X (gfx1250) — matrix operand order repaired and checked by reading, not compiled, not run on hardware
//
#include <hip/hip_runtime.h>

// ---------------------------------------------------------------------------
// Time-LSTM for MI455X (gfx1250, wave32, WMMA bf16 -> f32 accumulate)
//   B=128, SEQ=256, NC=16, HID=256, 4H=1024
//
// Fast path (ws_size permitting): precompute xp = x @ Uall^T for all steps
// (parallel, whole GPU), halving the sequential per-step WMMA work.
// xp is stored step-major [s][b][4H] so each recurrent step consumes one
// contiguous 32 KB slab per block (streamed + prefetched).
// Fallback: fully fused recurrence (K=512 = [h;x]).
// ---------------------------------------------------------------------------

#define B_    128
#define SEQ_  256
#define NC_   16
#define HID_  256
#define H4_   1024
#define XS_   264   // padded LDS row stride (ushorts): 528B = 33*16B -> b128-aligned, bank-rotating

typedef __attribute__((ext_vector_type(16))) __bf16        v16bf;
typedef __attribute__((ext_vector_type(8)))  float         v8f;
typedef __attribute__((ext_vector_type(8)))  unsigned int  v8ui;

__device__ __forceinline__ unsigned short f2bf(float x) {
  unsigned u = __float_as_uint(x);
  u += 0x7FFFu + ((u >> 16) & 1u);   // round-to-nearest-even
  return (unsigned short)(u >> 16);
}
__device__ __forceinline__ float b2f(unsigned short b) {
  return __uint_as_float(((unsigned)b) << 16);
}

// CDNA5 has a native V_TANH_F32 transcendental; use it when clang exposes it.
#if __has_builtin(__builtin_amdgcn_tanhf)
__device__ __forceinline__ float fast_tanh(float x) { return __builtin_amdgcn_tanhf(x); }
#else
__device__ __forceinline__ float fast_tanh(float x) { return tanhf(x); }
#endif
// sigmoid(x) = 0.5*tanh(x/2) + 0.5  -> 1 TRANS + 1 FMA
__device__ __forceinline__ float sigm(float x) {
  return __builtin_fmaf(0.5f, fast_tanh(0.5f * x), 0.5f);
}

// A-fragment (16x32 bf16, M=lane&15): lanes 0-15 -> K=kbase+{0..7,16..23}, kbase=0;
// lanes 16-31 -> kbase=8.  p = row_base + k0 + kbase; elems [0..7]=p[0..7], [8..15]=p[16..23].
__device__ __forceinline__ v16bf load_a_frag(const unsigned short* p) {
  v8ui t;
  const unsigned int* q0 = (const unsigned int*)p;
  const unsigned int* q1 = (const unsigned int*)(p + 16);
#pragma unroll
  for (int e = 0; e < 4; ++e) { t[e] = q0[e]; t[4 + e] = q1[e]; }
  return __builtin_bit_cast(v16bf, t);
}

// B-fragment (32x16 bf16, N=lane&15): lanes 0-15 -> K={0..7,16..23}, lanes 16-31 -> K={8..15,24..31}.
// p = W[n][k0 + bbase] for row-major [n][k] weights; elems [0..7]=p[0..7], [8..15]=p[16..23].
__device__ __forceinline__ v16bf load_b_frag(const unsigned short* p) {
  v8ui t;
  const unsigned int* q0 = (const unsigned int*)p;
  const unsigned int* q1 = (const unsigned int*)(p + 16);
#pragma unroll
  for (int e = 0; e < 4; ++e) { t[e] = q0[e]; t[4 + e] = q1[e]; }
  return __builtin_bit_cast(v16bf, t);
}

// ------------------------------- prep kernels ------------------------------

__global__ void cvt_bf16_kernel(const float* __restrict__ in,
                                unsigned short* __restrict__ out, int n) {
  int i = blockIdx.x * blockDim.x + threadIdx.x;
  if (i < n) out[i] = f2bf(in[i]);
}

__global__ void ts_kernel(const float* __restrict__ times,
                          float* __restrict__ tsv, int n) {
  int i = blockIdx.x * blockDim.x + threadIdx.x;
  if (i < n) tsv[i] = 1.0f / __logf(times[i] + 2.7183f);
}

// x[b,s,h] = sum_c emb[codes[b,s,c]][h] * mask[b,s,c]   -> bf16   (fallback path)
__global__ void embed_kernel(const int* __restrict__ codes,
                             const float* __restrict__ mask,
                             const float* __restrict__ emb,
                             unsigned short* __restrict__ xbf) {
  int bs = blockIdx.x;
  int h  = threadIdx.x;
  const int*   cd = codes + (size_t)bs * NC_;
  const float* mk = mask  + (size_t)bs * NC_;
  float acc = 0.f;
#pragma unroll
  for (int c = 0; c < NC_; ++c)
    acc += emb[(size_t)cd[c] * HID_ + h] * mk[c];
  xbf[(size_t)bs * HID_ + h] = f2bf(acc);
}

// ---------------- fast path: fused embed + xp = x @ Uall^T -----------------
// One block per 16-row M-tile of the flat (s*B + b) dimension (step-major!);
// 8 waves.  Phase 1: gather x tile (16x256) into LDS.  Phase 2: each wave
// does 8 N-tiles of the 16x1024x256 GEMM; store xp bf16 (bias added later).
__global__ __launch_bounds__(256)
void xp_fused_kernel(const int* __restrict__ codes,
                     const float* __restrict__ mask,
                     const float* __restrict__ emb,
                     const unsigned short* __restrict__ Uall,   // [4H, HID] bf16
                     unsigned short* __restrict__ xp) {         // [SEQ, B, 4H] bf16
  __shared__ unsigned short sh_xa[16 * XS_];

  const int tid = threadIdx.x;            // 0..255
  const int f0  = blockIdx.x * 16;        // flat step-major row: f = s*B + b
  const int s   = f0 / B_;                // tiles never straddle s (B=128, 8 tiles/s)
  const int bb  = f0 % B_;

  // phase 1: row j <-> batch bb+j at step s; emb rows are L2-resident (263 KB)
#pragma unroll 4
  for (int j = 0; j < 16; ++j) {
    const size_t bs = (size_t)(bb + j) * SEQ_ + s;
    const int*   cd = codes + bs * NC_;
    const float* mk = mask  + bs * NC_;
    float acc = 0.f;
#pragma unroll
    for (int c = 0; c < NC_; ++c)
      acc += emb[(size_t)cd[c] * HID_ + tid] * mk[c];
    sh_xa[j * XS_ + tid] = f2bf(acc);
  }
  __syncthreads();

  const int lane  = tid & 31;
  const int wave  = tid >> 5;
  const int r     = lane & 15;
  const int abase = (lane < 16) ? 0 : 8;
  const int bbase = (lane < 16) ? 0 : 8;
  const int mo    = (lane < 16) ? 0 : 8;

  for (int j = 0; j < 8; ++j) {
    const int n0 = (wave * 8 + j) * 16;      // N-tile origin in [0,1024)
    v8f acc = {};
#pragma unroll
    for (int ki = 0; ki < 8; ++ki) {
      const int k0 = ki * 32;
      v16bf a = load_a_frag(sh_xa + r * XS_ + k0 + abase);
      v16bf b = load_b_frag(Uall + (size_t)(n0 + r) * HID_ + k0 + bbase);
      acc = __builtin_amdgcn_wmma_f32_16x16x32_bf16(false, a, false, b, (short)0, acc, false, false);
    }
    const int gn = n0 + r;
#pragma unroll
    for (int v = 0; v < 8; ++v) {
      int m = v + mo;
      xp[(size_t)(f0 + m) * H4_ + gn] = f2bf(acc[v]);
    }
  }
}

// ---------------- fast path: recurrence (gate K = 256 only) ----------------
// waves 0-15 : gate GEMM, accumulators initialized from xp (step-major slab)
// waves 16-31: decay GEMM c_s1 = tanh(c @ Wd^T + b) -> c_adj
__global__ __launch_bounds__(1024)
void time_lstm_xp_kernel(const unsigned short* __restrict__ xp,   // [SEQ, B, 4H] bf16
                         const float* __restrict__ tsv,
                         const unsigned short* __restrict__ Wall, // [4H, HID] bf16
                         const unsigned short* __restrict__ Wd,   // [HID, HID] bf16
                         const float* __restrict__ Wall_b,
                         const float* __restrict__ Uall_b,
                         const float* __restrict__ Wd_b,
                         float* __restrict__ out) {
  __shared__ unsigned short sh_h[16 * XS_];
  __shared__ unsigned short sh_c16[16 * XS_];
  __shared__ float sh_c[16 * HID_];
  __shared__ float sh_cadj[16 * HID_];
  __shared__ float sh_t[16];

  const int tid  = threadIdx.x;
  const int lane = tid & 31;
  const int wave = tid >> 5;
  const int b0   = blockIdx.x * 16;

  for (int i = tid; i < 16 * XS_; i += 1024) { sh_h[i] = 0; sh_c16[i] = 0; }
  for (int i = tid; i < 16 * HID_; i += 1024) sh_c[i] = 0.f;
  __syncthreads();

  const int r     = lane & 15;
  const int abase = (lane < 16) ? 0 : 8;
  const int bbase = (lane < 16) ? 0 : 8;
  const int mo    = (lane < 16) ? 0 : 8;

  for (int s = 0; s < SEQ_; ++s) {
    if (tid < 16) sh_t[tid] = tsv[(size_t)(b0 + tid) * SEQ_ + s];
    // prefetch next step's contiguous 32 KB xp slab (global_prefetch_b8)
    if (s + 1 < SEQ_ && tid < 512) {
      const char* pf = (const char*)(xp + ((size_t)(s + 1) * B_ + b0) * H4_);
      __builtin_prefetch(pf + (size_t)tid * 64, 0, 3);
    }
    __syncthreads();

    v8f accF = {}, accI = {}, accO = {}, accC = {};
    v8f accD = {};

    if (wave < 16) {
      const int n0 = wave * 16;
      const int gn = n0 + r;
      // init accumulators from precomputed xp (x @ Uall^T); contiguous slab
      const unsigned short* xps = xp + ((size_t)s * B_ + b0) * H4_;
#pragma unroll
      for (int v = 0; v < 8; ++v) {
        const size_t row = (size_t)(v + mo) * H4_;
        accF[v] = b2f(xps[row + gn]);
        accI[v] = b2f(xps[row + 256 + gn]);
        accO[v] = b2f(xps[row + 512 + gn]);
        accC[v] = b2f(xps[row + 768 + gn]);
      }
      // h-part: K = 0..255 over Wall
#pragma unroll
      for (int ki = 0; ki < 8; ++ki) {
        const int k0 = ki * 32;
        v16bf a = load_a_frag(sh_h + r * XS_ + k0 + abase);
        const unsigned short* wp = Wall + (size_t)(n0 + r) * HID_ + k0 + bbase;
        v16bf bF = load_b_frag(wp);
        v16bf bI = load_b_frag(wp + (size_t)256 * HID_);
        v16bf bO = load_b_frag(wp + (size_t)512 * HID_);
        v16bf bC = load_b_frag(wp + (size_t)768 * HID_);
        accF = __builtin_amdgcn_wmma_f32_16x16x32_bf16(false, a, false, bF, (short)0, accF, false, false);
        accI = __builtin_amdgcn_wmma_f32_16x16x32_bf16(false, a, false, bI, (short)0, accI, false, false);
        accO = __builtin_amdgcn_wmma_f32_16x16x32_bf16(false, a, false, bO, (short)0, accO, false, false);
        accC = __builtin_amdgcn_wmma_f32_16x16x32_bf16(false, a, false, bC, (short)0, accC, false, false);
      }
    } else {
      const int n0 = (wave - 16) * 16;
#pragma unroll
      for (int ki = 0; ki < 8; ++ki) {
        const int k0 = ki * 32;
        v16bf a = load_a_frag(sh_c16 + r * XS_ + k0 + abase);
        v16bf b = load_b_frag(Wd + (size_t)(n0 + r) * HID_ + k0 + bbase);
        accD = __builtin_amdgcn_wmma_f32_16x16x32_bf16(false, a, false, b, (short)0, accD, false, false);
      }
      const int gn = n0 + r;
      const float bias = Wd_b[gn];
#pragma unroll
      for (int v = 0; v < 8; ++v) {
        int m = v + mo;
        float cs1  = fast_tanh(accD[v] + bias);
        float cold = sh_c[m * HID_ + gn];
        sh_cadj[m * HID_ + gn] = (cold - cs1) + cs1 * sh_t[m];
      }
    }
    __syncthreads();

    if (wave < 16) {
      const int n0 = wave * 16;
      const int gn = n0 + r;
      const float bf_ = Wall_b[gn]       + Uall_b[gn];
      const float bi_ = Wall_b[256 + gn] + Uall_b[256 + gn];
      const float bo_ = Wall_b[512 + gn] + Uall_b[512 + gn];
      const float bc_ = Wall_b[768 + gn] + Uall_b[768 + gn];
#pragma unroll
      for (int v = 0; v < 8; ++v) {
        int m = v + mo;
        float f  = sigm(accF[v] + bf_);
        float ii = sigm(accI[v] + bi_);
        float o  = sigm(accO[v] + bo_);
        float ct = sigm(accC[v] + bc_);
        float cn = __builtin_fmaf(f, sh_cadj[m * HID_ + gn], ii * ct);
        float hn = o * fast_tanh(cn);
        sh_c[m * HID_ + gn]  = cn;
        sh_c16[m * XS_ + gn] = f2bf(cn);
        sh_h[m * XS_ + gn]   = f2bf(hn);
        out[((size_t)(b0 + m) * SEQ_ + s) * HID_ + gn] = hn;
      }
    }
    __syncthreads();
  }
}

// -------------------- fallback: fully fused recurrence ---------------------
__global__ __launch_bounds__(1024)
void time_lstm_kernel(const unsigned short* __restrict__ xbf,   // [B*SEQ, HID] bf16
                      const float* __restrict__ tsv,
                      const unsigned short* __restrict__ Wall,
                      const unsigned short* __restrict__ Uall,
                      const unsigned short* __restrict__ Wd,
                      const float* __restrict__ Wall_b,
                      const float* __restrict__ Uall_b,
                      const float* __restrict__ Wd_b,
                      float* __restrict__ out) {
  __shared__ unsigned short sh_h[16 * XS_];
  __shared__ unsigned short sh_c16[16 * XS_];
  __shared__ unsigned short sh_x[16 * XS_];
  __shared__ float sh_c[16 * HID_];
  __shared__ float sh_cadj[16 * HID_];
  __shared__ float sh_t[16];

  const int tid  = threadIdx.x;
  const int lane = tid & 31;
  const int wave = tid >> 5;
  const int b0   = blockIdx.x * 16;

  for (int i = tid; i < 16 * XS_; i += 1024) { sh_h[i] = 0; sh_c16[i] = 0; }
  for (int i = tid; i < 16 * HID_; i += 1024) sh_c[i] = 0.f;
  __syncthreads();

  const int r     = lane & 15;
  const int abase = (lane < 16) ? 0 : 8;
  const int bbase = (lane < 16) ? 0 : 8;
  const int mo    = (lane < 16) ? 0 : 8;

  for (int s = 0; s < SEQ_; ++s) {
    {
      const unsigned int* src = (const unsigned int*)xbf;
      unsigned int* dst = (unsigned int*)sh_x;
      for (int i = tid; i < 16 * (HID_ / 2); i += 1024) {
        int m = i >> 7, kk = i & 127;
        dst[m * (XS_ / 2) + kk] =
            src[((size_t)(b0 + m) * SEQ_ + s) * (HID_ / 2) + kk];
      }
      if (tid < 16) sh_t[tid] = tsv[(size_t)(b0 + tid) * SEQ_ + s];
    }
    __syncthreads();

    v8f accF = {}, accI = {}, accO = {}, accC = {};
    v8f accD = {};

    if (wave < 16) {
      const int n0 = wave * 16;
#pragma unroll
      for (int ki = 0; ki < 8; ++ki) {
        const int k0 = ki * 32;
        v16bf a = load_a_frag(sh_h + r * XS_ + k0 + abase);
        const unsigned short* wp = Wall + (size_t)(n0 + r) * HID_ + k0 + bbase;
        v16bf bF = load_b_frag(wp);
        v16bf bI = load_b_frag(wp + (size_t)256 * HID_);
        v16bf bO = load_b_frag(wp + (size_t)512 * HID_);
        v16bf bC = load_b_frag(wp + (size_t)768 * HID_);
        accF = __builtin_amdgcn_wmma_f32_16x16x32_bf16(false, a, false, bF, (short)0, accF, false, false);
        accI = __builtin_amdgcn_wmma_f32_16x16x32_bf16(false, a, false, bI, (short)0, accI, false, false);
        accO = __builtin_amdgcn_wmma_f32_16x16x32_bf16(false, a, false, bO, (short)0, accO, false, false);
        accC = __builtin_amdgcn_wmma_f32_16x16x32_bf16(false, a, false, bC, (short)0, accC, false, false);
      }
#pragma unroll
      for (int ki = 0; ki < 8; ++ki) {
        const int k0 = ki * 32;
        v16bf a = load_a_frag(sh_x + r * XS_ + k0 + abase);
        const unsigned short* wp = Uall + (size_t)(n0 + r) * HID_ + k0 + bbase;
        v16bf bF = load_b_frag(wp);
        v16bf bI = load_b_frag(wp + (size_t)256 * HID_);
        v16bf bO = load_b_frag(wp + (size_t)512 * HID_);
        v16bf bC = load_b_frag(wp + (size_t)768 * HID_);
        accF = __builtin_amdgcn_wmma_f32_16x16x32_bf16(false, a, false, bF, (short)0, accF, false, false);
        accI = __builtin_amdgcn_wmma_f32_16x16x32_bf16(false, a, false, bI, (short)0, accI, false, false);
        accO = __builtin_amdgcn_wmma_f32_16x16x32_bf16(false, a, false, bO, (short)0, accO, false, false);
        accC = __builtin_amdgcn_wmma_f32_16x16x32_bf16(false, a, false, bC, (short)0, accC, false, false);
      }
    } else {
      const int n0 = (wave - 16) * 16;
#pragma unroll
      for (int ki = 0; ki < 8; ++ki) {
        const int k0 = ki * 32;
        v16bf a = load_a_frag(sh_c16 + r * XS_ + k0 + abase);
        v16bf b = load_b_frag(Wd + (size_t)(n0 + r) * HID_ + k0 + bbase);
        accD = __builtin_amdgcn_wmma_f32_16x16x32_bf16(false, a, false, b, (short)0, accD, false, false);
      }
      const int gn = n0 + r;
      const float bias = Wd_b[gn];
#pragma unroll
      for (int v = 0; v < 8; ++v) {
        int m = v + mo;
        float cs1  = fast_tanh(accD[v] + bias);
        float cold = sh_c[m * HID_ + gn];
        sh_cadj[m * HID_ + gn] = (cold - cs1) + cs1 * sh_t[m];
      }
    }
    __syncthreads();

    if (wave < 16) {
      const int n0 = wave * 16;
      const int gn = n0 + r;
      const float bf_ = Wall_b[gn]       + Uall_b[gn];
      const float bi_ = Wall_b[256 + gn] + Uall_b[256 + gn];
      const float bo_ = Wall_b[512 + gn] + Uall_b[512 + gn];
      const float bc_ = Wall_b[768 + gn] + Uall_b[768 + gn];
#pragma unroll
      for (int v = 0; v < 8; ++v) {
        int m = v + mo;
        float f  = sigm(accF[v] + bf_);
        float ii = sigm(accI[v] + bi_);
        float o  = sigm(accO[v] + bo_);
        float ct = sigm(accC[v] + bc_);
        float cn = __builtin_fmaf(f, sh_cadj[m * HID_ + gn], ii * ct);
        float hn = o * fast_tanh(cn);
        sh_c[m * HID_ + gn]  = cn;
        sh_c16[m * XS_ + gn] = f2bf(cn);
        sh_h[m * XS_ + gn]   = f2bf(hn);
        out[((size_t)(b0 + m) * SEQ_ + s) * HID_ + gn] = hn;
      }
    }
    __syncthreads();
  }
}

// ------------------------------- launcher ----------------------------------

extern "C" void kernel_launch(void* const* d_in, const int* in_sizes, int n_in,
                              void* d_out, int out_size, void* d_ws, size_t ws_size,
                              hipStream_t stream) {
  (void)in_sizes; (void)n_in; (void)out_size;

  const int*   codes  = (const int*)  d_in[0];
  const float* mask   = (const float*)d_in[1];
  const float* times  = (const float*)d_in[2];
  const float* emb    = (const float*)d_in[3];
  const float* Wall_w = (const float*)d_in[4];
  const float* Wall_b = (const float*)d_in[5];
  const float* Uall_w = (const float*)d_in[6];
  const float* Uall_b = (const float*)d_in[7];
  const float* Wd_w   = (const float*)d_in[8];
  const float* Wd_b   = (const float*)d_in[9];
  float* out = (float*)d_out;

  const size_t szXP   = (size_t)B_ * SEQ_ * H4_ * 2;   // 64 MB (bf16 xp)
  const size_t szXBF  = (size_t)B_ * SEQ_ * HID_ * 2;  // 16 MB (bf16 x)
  const size_t szWall = (size_t)H4_ * HID_ * 2;
  const size_t szWd   = (size_t)HID_ * HID_ * 2;
  const size_t szTs   = (size_t)B_ * SEQ_ * 4;
  const size_t needFast = szXP + 2 * szWall + szWd + szTs;      // ~68.4 MB
  const bool fast = (ws_size >= needFast);

  char* ws = (char*)d_ws;
  if (fast) {
    unsigned short* xp    = (unsigned short*)ws; ws += szXP;
    unsigned short* WallB = (unsigned short*)ws; ws += szWall;
    unsigned short* UallB = (unsigned short*)ws; ws += szWall;
    unsigned short* WdB   = (unsigned short*)ws; ws += szWd;
    float*          tsv   = (float*)ws;

    cvt_bf16_kernel<<<(H4_ * HID_ + 255) / 256, 256, 0, stream>>>(Wall_w, WallB, H4_ * HID_);
    cvt_bf16_kernel<<<(H4_ * HID_ + 255) / 256, 256, 0, stream>>>(Uall_w, UallB, H4_ * HID_);
    cvt_bf16_kernel<<<(HID_ * HID_ + 255) / 256, 256, 0, stream>>>(Wd_w, WdB, HID_ * HID_);
    ts_kernel<<<(B_ * SEQ_ + 255) / 256, 256, 0, stream>>>(times, tsv, B_ * SEQ_);
    xp_fused_kernel<<<(B_ * SEQ_) / 16, 256, 0, stream>>>(codes, mask, emb, UallB, xp);
    time_lstm_xp_kernel<<<B_ / 16, 1024, 0, stream>>>(xp, tsv, WallB, WdB,
                                                      Wall_b, Uall_b, Wd_b, out);
  } else {
    unsigned short* xbf   = (unsigned short*)ws; ws += szXBF;
    unsigned short* WallB = (unsigned short*)ws; ws += szWall;
    unsigned short* UallB = (unsigned short*)ws; ws += szWall;
    unsigned short* WdB   = (unsigned short*)ws; ws += szWd;
    float*          tsv   = (float*)ws;

    cvt_bf16_kernel<<<(H4_ * HID_ + 255) / 256, 256, 0, stream>>>(Wall_w, WallB, H4_ * HID_);
    cvt_bf16_kernel<<<(H4_ * HID_ + 255) / 256, 256, 0, stream>>>(Uall_w, UallB, H4_ * HID_);
    cvt_bf16_kernel<<<(HID_ * HID_ + 255) / 256, 256, 0, stream>>>(Wd_w, WdB, HID_ * HID_);
    ts_kernel<<<(B_ * SEQ_ + 255) / 256, 256, 0, stream>>>(times, tsv, B_ * SEQ_);
    embed_kernel<<<B_ * SEQ_, HID_, 0, stream>>>(codes, mask, emb, xbf);
    time_lstm_kernel<<<B_ / 16, 1024, 0, stream>>>(xbf, tsv, WallB, UallB, WdB,
                                                   Wall_b, Uall_b, Wd_b, out);
  }
}
